// MultiScaleInstanceSegmentorTTA_76355928588770
// MI455X (gfx1250) — 8-bit matrix operand order repaired and checked by reading, compiled, not run on hardware
//
#include <hip/hip_runtime.h>
#include <hip/hip_bf16.h>


#define N_DET    256
#define H_IMG    512
#define W_IMG    512
#define HW       (H_IMG * W_IMG)
#define MAX_DET  100
#define KSLICES  32
#define NMS_THR  0.5f
#define FP8_ONE  0x38

typedef __attribute__((ext_vector_type(16))) int   v16i;
typedef __attribute__((ext_vector_type(8)))  float v8f;

__global__ __launch_bounds__(256) void nmszero_kernel(float* __restrict__ p, int n) {
    int i = blockIdx.x * blockDim.x + threadIdx.x;
    if (i < n) p[i] = 0.0f;
}

__global__ __launch_bounds__(256) void nmspack_kernel(const float* __restrict__ m,
                                                      unsigned char* __restrict__ pk,
                                                      int total4) {
    int idx = blockIdx.x * blockDim.x + threadIdx.x;
    if (idx >= total4) return;
    float4 v = ((const float4*)m)[idx];
    uchar4 o;
    o.x = v.x > 0.5f ? FP8_ONE : 0;
    o.y = v.y > 0.5f ? FP8_ONE : 0;
    o.z = v.z > 0.5f ? FP8_ONE : 0;
    o.w = v.w > 0.5f ? FP8_ONE : 0;
    ((uchar4*)pk)[idx] = o;
}

__global__ __launch_bounds__(32) void nmsgram_wmma_kernel(
        const unsigned char* __restrict__ Apk, float* __restrict__ inter) {
    const int ti   = blockIdx.x;
    const int tj   = blockIdx.y;
    const int ks   = blockIdx.z;
    const int lane = threadIdx.x;
    const int half = lane >> 4;
    const int mn   = lane & 15;

    const int kPerSlice = HW / KSLICES;
    const int k0base    = ks * kPerSlice;

    const unsigned char* aRow = Apk + (size_t)(ti * 16 + mn) * HW + k0base;
    const unsigned char* bRow = Apk + (size_t)(tj * 16 + mn) * HW + k0base;

    v8f acc = {0.f, 0.f, 0.f, 0.f, 0.f, 0.f, 0.f, 0.f};

    #pragma unroll 2
    for (int kk = 0; kk < kPerSlice; kk += 128) {
        const unsigned char* ap = aRow + kk;
        const unsigned char* bp = bRow + kk;
        v16i a, b;
        a[0]  = *(const int*)(ap +  0 + 8 * half + 0);
        a[1]  = *(const int*)(ap +  0 + 8 * half + 4);
        a[2]  = *(const int*)(ap + 16 + 8 * half + 0);
        a[3]  = *(const int*)(ap + 16 + 8 * half + 4);
        a[4]  = *(const int*)(ap + 32 + 8 * half + 0);
        a[5]  = *(const int*)(ap + 32 + 8 * half + 4);
        a[6]  = *(const int*)(ap + 48 + 8 * half + 0);
        a[7]  = *(const int*)(ap + 48 + 8 * half + 4);
        a[8]  = *(const int*)(ap + 64 + 8 * half + 0);
        a[9]  = *(const int*)(ap + 64 + 8 * half + 4);
        a[10] = *(const int*)(ap + 80 + 8 * half + 0);
        a[11] = *(const int*)(ap + 80 + 8 * half + 4);
        a[12] = *(const int*)(ap + 96 + 8 * half + 0);
        a[13] = *(const int*)(ap + 96 + 8 * half + 4);
        a[14] = *(const int*)(ap + 112 + 8 * half + 0);
        a[15] = *(const int*)(ap + 112 + 8 * half + 4);
        b[0]  = *(const int*)(bp +  0 + 8 * half + 0);
        b[1]  = *(const int*)(bp +  0 + 8 * half + 4);
        b[2]  = *(const int*)(bp + 16 + 8 * half + 0);
        b[3]  = *(const int*)(bp + 16 + 8 * half + 4);
        b[4]  = *(const int*)(bp + 32 + 8 * half + 0);
        b[5]  = *(const int*)(bp + 32 + 8 * half + 4);
        b[6]  = *(const int*)(bp + 48 + 8 * half + 0);
        b[7]  = *(const int*)(bp + 48 + 8 * half + 4);
        b[8]  = *(const int*)(bp + 64 + 8 * half + 0);
        b[9]  = *(const int*)(bp + 64 + 8 * half + 4);
        b[10] = *(const int*)(bp + 80 + 8 * half + 0);
        b[11] = *(const int*)(bp + 80 + 8 * half + 4);
        b[12] = *(const int*)(bp + 96 + 8 * half + 0);
        b[13] = *(const int*)(bp + 96 + 8 * half + 4);
        b[14] = *(const int*)(bp + 112 + 8 * half + 0);
        b[15] = *(const int*)(bp + 112 + 8 * half + 4);
        acc = __builtin_amdgcn_wmma_f32_16x16x128_fp8_fp8(a, b, (short)0, acc,
                                                          false, false);
    }

    #pragma unroll
    for (int r = 0; r < 8; ++r) {
        int M = ti * 16 + r + 8 * half;
        int N = tj * 16 + mn;
        atomicAdd(&inter[M * N_DET + N], acc[r]);
    }
}

__global__ __launch_bounds__(256) void nmsgreedy_kernel(
        const float* __restrict__ inter,
        const float* __restrict__ scores,
        const int* __restrict__ classes,
        int* __restrict__ keep_idx, int* __restrict__ keep_valid,
        float* __restrict__ out_scores, float* __restrict__ out_classes,
        float* __restrict__ out_valid) {
    __shared__ float s_score[N_DET];
    __shared__ int   s_order[N_DET];
    __shared__ int   s_ocls[N_DET];
    __shared__ float s_oarea[N_DET];
    __shared__ int   s_alive[N_DET];
    __shared__ int   s_keep[MAX_DET];
    __shared__ int   s_val[MAX_DET];

    const int tid = threadIdx.x;

    s_score[tid] = scores[tid];
    __syncthreads();

    const float si = s_score[tid];
    int rank = 0;
    for (int j = 0; j < N_DET; ++j) {
        float sj = s_score[j];
        rank += (sj > si) || (sj == si && j < tid);
    }
    s_order[rank] = tid;
    __syncthreads();

    const int oi = s_order[tid];
    s_ocls[tid]  = classes[oi];
    s_oarea[tid] = inter[oi * N_DET + oi];
    s_alive[tid] = 1;
    if (tid < MAX_DET) { s_keep[tid] = 0; s_val[tid] = 0; }
    __syncthreads();

    for (int p = 0; p < N_DET; ++p) {
        if (tid > p && s_alive[p] && s_alive[tid] &&
            s_ocls[p] == s_ocls[tid]) {
            int a = s_order[p];
            int b = s_order[tid];
            float I = inter[a * N_DET + b];
            float U = s_oarea[p] + s_oarea[tid] - I;
            if (I / (U + 1e-6f) > NMS_THR) s_alive[tid] = 0;
        }
        __syncthreads();
    }

    int pos = 0;
    for (int k = 0; k < tid; ++k) pos += s_alive[k];
    if (s_alive[tid] && pos < MAX_DET) {
        s_keep[pos] = s_order[tid];
        s_val[pos]  = 1;
    }
    __syncthreads();

    if (tid < MAX_DET) {
        int v  = s_val[tid];
        int ki = s_keep[tid];
        keep_idx[tid]   = ki;
        keep_valid[tid] = v;
        out_scores[tid]  = v ? scores[ki] : 0.0f;
        out_classes[tid] = v ? (float)classes[ki] : -1.0f;
        out_valid[tid]   = (float)v;
    }
}

__global__ __launch_bounds__(256) void nmsgather_kernel(
        const float* __restrict__ masks,
        const int* __restrict__ keep_idx, const int* __restrict__ keep_valid,
        float* __restrict__ out_masks) {
    const int slot = blockIdx.y;
    const int idx  = blockIdx.x * blockDim.x + threadIdx.x;
    float4 v = make_float4(0.f, 0.f, 0.f, 0.f);
    if (keep_valid[slot]) {
        v = ((const float4*)(masks + (size_t)keep_idx[slot] * HW))[idx];
    }
    ((float4*)(out_masks + (size_t)slot * HW))[idx] = v;
}

__global__ __launch_bounds__(256) void nmsboxes_kernel(
        const unsigned char* __restrict__ pk,
        const int* __restrict__ keep_idx, const int* __restrict__ keep_valid,
        float* __restrict__ out_boxes) {
    const int slot = blockIdx.x;
    const int tid  = threadIdx.x;

    __shared__ int gminx, gminy, gmaxx, gmaxy;
    if (tid == 0) { gminx = 1 << 30; gminy = 1 << 30; gmaxx = -1; gmaxy = -1; }
    __syncthreads();

    int minx = 1 << 30, miny = 1 << 30, maxx = -1, maxy = -1;
    if (keep_valid[slot]) {
        const unsigned char* row = pk + (size_t)keep_idx[slot] * HW;
        const uint4* r16 = (const uint4*)row;
        for (int c = tid; c < HW / 16; c += 256) {
            uint4 v = r16[c];
            if (v.x | v.y | v.z | v.w) {
                int base = c * 16;
                #pragma unroll
                for (int b = 0; b < 16; ++b) {
                    if (row[base + b]) {
                        int p = base + b;
                        int x = p & (W_IMG - 1);
                        int y = p >> 9;
                        minx = min(minx, x); maxx = max(maxx, x);
                        miny = min(miny, y); maxy = max(maxy, y);
                    }
                }
            }
        }
    }
    atomicMin(&gminx, minx); atomicMax(&gmaxx, maxx);
    atomicMin(&gminy, miny); atomicMax(&gmaxy, maxy);
    __syncthreads();

    if (tid == 0) {
        float4 box = make_float4(0.f, 0.f, 0.f, 0.f);
        if (gmaxx >= 0) {
            box = make_float4((float)gminx, (float)gminy,
                              (float)(gmaxx + 1), (float)(gmaxy + 1));
        }
        ((float4*)out_boxes)[slot] = box;
    }
}

extern "C" void kernel_launch(void* const* d_in, const int* in_sizes, int n_in,
                              void* d_out, int out_size, void* d_ws, size_t ws_size,
                              hipStream_t stream) {
    (void)in_sizes; (void)n_in; (void)out_size; (void)ws_size;

    const float* masks   = (const float*)d_in[0];
    const float* scores  = (const float*)d_in[1];
    const int*   classes = (const int*)d_in[2];

    unsigned char* pk   = (unsigned char*)d_ws;
    float* inter        = (float*)((char*)d_ws + (size_t)N_DET * HW);
    int* keep_idx       = (int*)(inter + N_DET * N_DET);
    int* keep_valid     = keep_idx + 128;

    float* out        = (float*)d_out;
    float* out_masks  = out;
    float* out_boxes  = out_masks + (size_t)MAX_DET * HW;
    float* out_scores = out_boxes + MAX_DET * 4;
    float* out_class  = out_scores + MAX_DET;
    float* out_valid  = out_class + MAX_DET;

    nmszero_kernel<<<dim3((N_DET * N_DET + 255) / 256), 256, 0, stream>>>(
        inter, N_DET * N_DET);

    const int total4 = N_DET * HW / 4;
    nmspack_kernel<<<dim3(total4 / 256), 256, 0, stream>>>(masks, pk, total4);

    nmsgram_wmma_kernel<<<dim3(16, 16, KSLICES), 32, 0, stream>>>(pk, inter);

    nmsgreedy_kernel<<<1, 256, 0, stream>>>(inter, scores, classes,
                                            keep_idx, keep_valid,
                                            out_scores, out_class, out_valid);

    nmsgather_kernel<<<dim3(HW / 4 / 256, MAX_DET), 256, 0, stream>>>(
        masks, keep_idx, keep_valid, out_masks);

    nmsboxes_kernel<<<dim3(MAX_DET), 256, 0, stream>>>(pk, keep_idx, keep_valid,
                                                       out_boxes);
}
